// FusionOnManifold_9560597201095
// MI455X (gfx1250) — matrix operand order repaired and checked by reading, not compiled, not run on hardware
//
#include <hip/hip_runtime.h>
#include <hip/hip_bf16.h>
#include <math.h>

// ---------------------------------------------------------------------------
// FusionOnManifold for MI455X (gfx1250, wave32).
//   GEMM1 (16384x2048 @ 2048x257) and GEMM2 (16384x257 @ 257x2048) on
//   v_wmma_f32_16x16x32_bf16; manifold math f32 VALU with wave32 shuffle
//   reductions. 32 rows per workgroup (2 M-subtiles per wave) so every B
//   fragment feeds two WMMAs -> halves L2 B-traffic vs a 16-row tile.
//   HBM traffic: e_out + y_h in, z_fused + alpha out (~275 MB ~ 12us floor).
// ---------------------------------------------------------------------------

typedef __attribute__((ext_vector_type(16))) __bf16 v16bf;
typedef __attribute__((ext_vector_type(8)))  __bf16 v8bf;
typedef __attribute__((ext_vector_type(8)))  float  v8f;

#define EPSF    1e-12f
#define MAXSTEP 5.0f
#define MAXRAD  10.0f

#define DDIM 2048          // D
#define ADIM 257           // A
#define APAD 272           // A padded to mult of 16 (GEMM1 N)
#define KPAD 288           // A padded to mult of 32 (GEMM2 K)
#define ROWS 16384         // B*T
#define MTILE 32           // rows per workgroup
#define EKC 128            // e_out K staging chunk
#define EAS 136            // eA row stride (halves): 272B, 16B-aligned, bank-spread
#define ZTS 296            // zt row stride (halves): 592B, 16B-aligned

static __device__ __forceinline__ unsigned short f2bf(float x) {
  __bf16 b = (__bf16)x;
  return __builtin_bit_cast(unsigned short, b);
}

static __device__ __forceinline__ v16bf mk16(uint4 lo, uint4 hi) {
  v8bf l = __builtin_bit_cast(v8bf, lo);
  v8bf h = __builtin_bit_cast(v8bf, hi);
  return __builtin_shufflevector(l, h, 0,1,2,3,4,5,6,7,8,9,10,11,12,13,14,15);
}

static __device__ __forceinline__ v8f wmma_bf16(v16bf a, v16bf b, v8f c) {
  return __builtin_amdgcn_wmma_f32_16x16x32_bf16(false, a, false, b,
                                                 (short)0, c, false, false);
}

static __device__ __forceinline__ float wred(float x) {
#pragma unroll
  for (int m = 16; m > 0; m >>= 1) x += __shfl_xor(x, m, 32);
  return x;
}

// Minkowski inner product of two 257-vectors striped across a wave32:
// lane L holds elements L, L+32, ... (9 slots, zero past 257).
static __device__ __forceinline__ float mink9(const float* a, const float* b) {
  float t = 0.f;
#pragma unroll
  for (int k = 0; k < 9; ++k) t += a[k] * b[k];
  t = wred(t);
  float ab0 = __shfl(a[0] * b[0], 0, 32);
  return t - 2.f * ab0;
}

// ---------------------------------------------------------------------------
// Kernel 0: zero-padded bf16 weight copies in workspace.
//   we: [272][2048] (W_e^T, rows >=257 zero) ; wo: [2048][288] (W_o^T, cols >=257 zero)
// ---------------------------------------------------------------------------
__global__ __launch_bounds__(256) void fom_cvt_weights(
    const float* __restrict__ We, const float* __restrict__ Wo,
    unsigned short* __restrict__ we, unsigned short* __restrict__ wo) {
  int idx = blockIdx.x * 256 + threadIdx.x;
  const int NWE = DDIM * APAD;
  const int NWO = KPAD * DDIM;
  if (idx < NWE) {
    int row = idx / APAD, col = idx - row * APAD;
    we[col * DDIM + row] = (col < ADIM) ? f2bf(We[row * ADIM + col]) : (unsigned short)0;
  } else if (idx < NWE + NWO) {
    int j = idx - NWE;
    int row = j >> 11, col = j & (DDIM - 1);
    wo[col * KPAD + row] = (row < ADIM) ? f2bf(Wo[row * DDIM + col]) : (unsigned short)0;
  }
}

// ---------------------------------------------------------------------------
// Fused kernel: one workgroup (8 waves, 256 thr) per 32-row tile.
// ---------------------------------------------------------------------------
__global__ __launch_bounds__(256) void fom_fused(
    const float* __restrict__ e_out, const float* __restrict__ y_h,
    const float* __restrict__ z_anchor, const float* __restrict__ z_gate,
    const float* __restrict__ a_gate, const float* __restrict__ score_scale,
    const float* __restrict__ b_e, const float* __restrict__ b_o,
    const unsigned short* __restrict__ we, const unsigned short* __restrict__ wo,
    float* __restrict__ out) {
  // Static LDS ~62.6 KB
  __shared__ unsigned short eA[MTILE * EAS];   // e_out K-chunk, bf16
  __shared__ float          v_sh[MTILE * ADIM];// GEMM1 result + b_e, f32
  __shared__ unsigned short zt[MTILE * ZTS];   // z_tan bf16, K zero-padded
  __shared__ float          p_sh[257];
  __shared__ float          zg_sh[256];
  __shared__ float          sc_p0, sc_sdr, sc_zn;

  const int tid  = threadIdx.x;
  const int lane = tid & 31;
  const int wave = tid >> 5;
  const int wave_s = __builtin_amdgcn_readfirstlane(wave); // uniform -> scalar branches
  const int r0   = blockIdx.x * MTILE;
  const int m    = lane & 15;          // M (A frag) / N (C frag) lane role
  const int koff = (lane >> 4) << 3;   // 0 or 8: A-fragment K offset per half-wave
  const int rbase = (lane >> 4) << 3;  // C fragment row base

  // ---------------- Phase A: z_gate, anchor point p ------------------------
  for (int i = tid; i < MTILE * ZTS; i += 256) zt[i] = 0;
  zg_sh[tid] = z_gate[tid];
  if (wave_s == 0) {
    float s1 = 0.f, s2 = 0.f;
    for (int i = lane; i < 256; i += 32) {
      float za = z_anchor[i]; s1 += za * za;
      float zg = z_gate[i];   s2 += zg * zg;
    }
    s1 = wred(s1); s2 = wred(s2);
    if (lane == 0) {
      float rn = sqrtf(s1 + EPSF);
      float rc = fminf(rn, MAXRAD);
      sc_p0  = coshf(rc);
      sc_sdr = sinhf(rc) / rn;
      sc_zn  = sqrtf(s2 + EPSF);
    }
  }
  __syncthreads();
  if (tid == 0) p_sh[0] = sc_p0;
  p_sh[1 + tid] = sc_sdr * z_anchor[tid];
  // (p_sh consumed only after GEMM1's barriers)

  // ---------------- GEMM1: v = e_out @ W_e  (bf16 WMMA) --------------------
  // 17 N-tiles over 8 waves; wave 0 also owns tile 16 (scalar-uniform branch).
  const int nt0 = wave, nt1 = wave + 8;
  const bool third = (wave_s == 0);
  v8f c00, c01, c10, c11, c20, c21;
#pragma unroll
  for (int e = 0; e < 8; ++e) {
    c00[e] = 0.f; c01[e] = 0.f; c10[e] = 0.f;
    c11[e] = 0.f; c20[e] = 0.f; c21[e] = 0.f;
  }

  for (int kc0 = 0; kc0 < DDIM; kc0 += EKC) {
    __syncthreads();
    // cooperative load+convert of 32x128 e_out chunk (1024 float4)
    for (int idx = tid; idx < 1024; idx += 256) {
      int row = idx >> 5, c4 = idx & 31;
      const float4 v4 =
          *(const float4*)&e_out[(size_t)(r0 + row) * DDIM + kc0 + c4 * 4];
      if (kc0 + EKC < DDIM)
        __builtin_prefetch(&e_out[(size_t)(r0 + row) * DDIM + kc0 + EKC + c4 * 4], 0, 1);
      unsigned short* d = &eA[row * EAS + c4 * 4];
      d[0] = f2bf(v4.x); d[1] = f2bf(v4.y); d[2] = f2bf(v4.z); d[3] = f2bf(v4.w);
    }
    __syncthreads();
    for (int kk = 0; kk < EKC; kk += 32) {
      // A fragments for the wave's two M-subtiles (lane=M, halves=K pairs)
      const uint4* ap0 = (const uint4*)&eA[m * EAS + kk + koff];
      const uint4* ap1 = (const uint4*)&eA[(m + 16) * EAS + kk + koff];
      v16bf a0 = mk16(ap0[0], ap0[2]);
      v16bf a1 = mk16(ap1[0], ap1[2]);
      int krow = kc0 + kk + koff;          // B fragment: lane=N, halves=K pairs
      const uint4* bp0 = (const uint4*)&we[((nt0 << 4) + m) * DDIM + krow];
      v16bf b0 = mk16(bp0[0], bp0[2]);
      c00 = wmma_bf16(a0, b0, c00);
      c01 = wmma_bf16(a1, b0, c01);
      const uint4* bp1 = (const uint4*)&we[((nt1 << 4) + m) * DDIM + krow];
      v16bf b1 = mk16(bp1[0], bp1[2]);
      c10 = wmma_bf16(a0, b1, c10);
      c11 = wmma_bf16(a1, b1, c11);
      if (third) {                          // scalar-uniform: EXEC untouched
        const uint4* bp2 = (const uint4*)&we[((16 << 4) + m) * DDIM + krow];
        v16bf b2 = mk16(bp2[0], bp2[2]);
        c20 = wmma_bf16(a0, b2, c20);
        c21 = wmma_bf16(a1, b2, c21);
      }
    }
  }
  __syncthreads();
  {
    // C layout: VGPR j, lane l -> D[M = j + 8*(l>=16), N = l&15]
    auto store_pair = [&](const v8f& lo, const v8f& hi, int ntv) {
      int col = (ntv << 4) + m;
      if (col < ADIM) {
        float be = b_e[col];
#pragma unroll
        for (int j = 0; j < 8; ++j) {
          v_sh[(rbase + j) * ADIM + col]        = lo[j] + be;
          v_sh[(16 + rbase + j) * ADIM + col]   = hi[j] + be;
        }
      }
    };
    store_pair(c00, c01, nt0);
    store_pair(c10, c11, nt1);
    if (third) store_pair(c20, c21, 16);
  }
  __syncthreads();

  // ---------------- Phase B: manifold math, 4 rows per wave ----------------
  const float ag = fminf(fmaxf(a_gate[0], -MAXRAD), MAXRAD);
  const float ss = score_scale[0];
  const float ca = coshf(ag), sa = sinhf(ag);
  const float zn = sc_zn;

  for (int rsel = 0; rsel < 4; ++rsel) {
    const int rr = wave * 4 + rsel;
    float pv[9], yv[9], vv[9];
#pragma unroll
    for (int k = 0; k < 9; ++k) {
      int i = lane + (k << 5);
      bool ok = i < ADIM;
      pv[k] = ok ? p_sh[i] : 0.f;
      yv[k] = ok ? y_h[(size_t)(r0 + rr) * ADIM + i] : 0.f;  // coalesced
      vv[k] = ok ? v_sh[rr * ADIM + i] : 0.f;
    }
    // gate: s = cosh(a)<y_spatial,z_gate> - sinh(a)*||z||*y0
    float sg = 0.f;
#pragma unroll
    for (int k = 0; k < 9; ++k) {
      int i = lane + (k << 5);
      if (i >= 1 && i < ADIM) sg += yv[k] * zg_sh[i - 1];
    }
    sg = wred(sg);
    float y0 = __shfl(yv[0], 0, 32);
    float s  = ca * sg - sa * zn * y0;
    float alpha = 1.f / (1.f + expf(-ss * s));
    alpha = fminf(fmaxf(alpha, 1e-6f), 1.f - 1e-6f);
    if (lane == 0) out[(size_t)ROWS * DDIM + (r0 + rr)] = alpha;

    // project to T_p, clamp, exp to manifold
    float b1 = mink9(pv, vv);
#pragma unroll
    for (int k = 0; k < 9; ++k) vv[k] = fmaf(b1, pv[k], vv[k]);
    float q   = mink9(vv, vv);
    float vn  = sqrtf(fmaxf(fabsf(q), EPSF));
    float scv = fminf(MAXSTEP / vn, 1.f);
#pragma unroll
    for (int k = 0; k < 9; ++k) vv[k] *= scv;
    float nv  = sqrtf(fmaxf(fabsf(q) * scv * scv, EPSF));
    float chv = coshf(nv), shv = sinhf(nv) / nv;
    float xe[9];
#pragma unroll
    for (int k = 0; k < 9; ++k) xe[k] = chv * pv[k] + shv * vv[k];

    // geodesic interpolation toward y
    float beta = mink9(xe, yv);
    float dd   = acoshf(fmaxf(-beta, 1.f + 1e-7f));
    float uu[9];
#pragma unroll
    for (int k = 0; k < 9; ++k) uu[k] = fmaf(beta, xe[k], yv[k]);
    float qu  = mink9(uu, uu);
    float aqu = fabsf(qu);
    float un  = sqrtf(fmaxf(aqu, EPSF));
    float lg  = dd / un;
    float un2 = sqrtf(fmaxf(lg * lg * aqu, EPSF));
    float fs  = alpha * lg * fminf(MAXSTEP / un2, 1.f);
    float ns  = sqrtf(fmaxf(fs * fs * aqu, EPSF));
    float ch2 = coshf(ns), sh2 = sinhf(ns) * (fs / ns);
    float zm[9];
#pragma unroll
    for (int k = 0; k < 9; ++k) zm[k] = ch2 * xe[k] + sh2 * uu[k];

    // log back at p
    float b2  = mink9(pv, zm);
    float d2  = acoshf(fmaxf(-b2, 1.f + 1e-7f));
    float u2[9];
#pragma unroll
    for (int k = 0; k < 9; ++k) u2[k] = fmaf(b2, pv[k], zm[k]);
    float qu2 = mink9(u2, u2);
    float un3 = sqrtf(fmaxf(fabsf(qu2), EPSF));
    float g   = d2 / un3;
#pragma unroll
    for (int k = 0; k < 9; ++k) {
      int i = lane + (k << 5);
      if (i < ADIM) zt[rr * ZTS + i] = f2bf(g * u2[k]);
    }
  }
  __syncthreads();

  // ---------------- GEMM2: z_fused = z_tan @ W_o + b_o ---------------------
  for (int j = 0; j < 16; ++j) {            // 128 N-tiles / 8 waves
    int n0 = (wave + (j << 3)) << 4;
    v8f d0, d1;
#pragma unroll
    for (int e = 0; e < 8; ++e) { d0[e] = 0.f; d1[e] = 0.f; }
#pragma unroll
    for (int kc = 0; kc < KPAD; kc += 32) {
      const uint4* ap0 = (const uint4*)&zt[m * ZTS + kc + koff];
      const uint4* ap1 = (const uint4*)&zt[(m + 16) * ZTS + kc + koff];
      v16bf a0 = mk16(ap0[0], ap0[2]);
      v16bf a1 = mk16(ap1[0], ap1[2]);
      const uint4* bp = (const uint4*)&wo[(n0 + m) * KPAD + kc + koff];
      v16bf b = mk16(bp[0], bp[2]);         // one B fragment feeds 2 WMMAs
      d0 = wmma_bf16(a0, b, d0);
      d1 = wmma_bf16(a1, b, d1);
    }
    int col = n0 + m;
    float bo = b_o[col];
#pragma unroll
    for (int jj = 0; jj < 8; ++jj) {
      out[(size_t)(r0 + rbase + jj) * DDIM + col]      = d0[jj] + bo;
      out[(size_t)(r0 + 16 + rbase + jj) * DDIM + col] = d1[jj] + bo;
    }
  }
}

// ---------------------------------------------------------------------------
extern "C" void kernel_launch(void* const* d_in, const int* in_sizes, int n_in,
                              void* d_out, int out_size, void* d_ws, size_t ws_size,
                              hipStream_t stream) {
  (void)in_sizes; (void)n_in; (void)out_size; (void)ws_size;
  const float* e_out       = (const float*)d_in[0];
  const float* y_h         = (const float*)d_in[1];
  const float* z_anchor    = (const float*)d_in[2];
  const float* z_gate      = (const float*)d_in[3];
  const float* a_gate      = (const float*)d_in[4];
  const float* score_scale = (const float*)d_in[5];
  const float* W_e         = (const float*)d_in[6];
  const float* b_e         = (const float*)d_in[7];
  const float* W_o         = (const float*)d_in[8];
  const float* b_o         = (const float*)d_in[9];
  float* out = (float*)d_out;

  unsigned short* we = (unsigned short*)d_ws;            // 2048*272 bf16
  unsigned short* wo = we + DDIM * APAD;                 // 288*2048 bf16

  const int total = DDIM * APAD + KPAD * DDIM;
  fom_cvt_weights<<<(total + 255) / 256, 256, 0, stream>>>(W_e, W_o, we, wo);
  fom_fused<<<ROWS / MTILE, 256, 0, stream>>>(e_out, y_h, z_anchor, z_gate,
                                              a_gate, score_scale, b_e, b_o,
                                              we, wo, out);
}
